// BertSelfAttention_28802050687766
// MI455X (gfx1250) — compile-verified, matrix operand order checked by reading, not run on hardware
//
#include <hip/hip_runtime.h>
#include <stddef.h>
#include <stdint.h>

// ---------------- types ----------------
typedef __attribute__((ext_vector_type(16))) __bf16 v16bf;
typedef __attribute__((ext_vector_type(8)))  __bf16 v8bf;
typedef __attribute__((ext_vector_type(8)))  float  v8f;

#define Bsz 4
#define Ssz 2048
#define DMsz 1024
#define Hsz 16
#define Dsz 64
#define L2E 1.4426950408889634f

#define WAIT_ASYNCCNT(n) asm volatile("s_wait_asynccnt " #n ::: "memory")
#define WAIT_DSCNT0      asm volatile("s_wait_dscnt 0x0" ::: "memory")

// fp32 -> bf16 round-to-nearest-even
__device__ __forceinline__ __bf16 f2bf(float f) {
  union { float f; unsigned u; } a; a.f = f;
  unsigned r = a.u + 0x7FFFu + ((a.u >> 16) & 1u);
  union { unsigned short s; __bf16 b; } o; o.s = (unsigned short)(r >> 16);
  return o.b;
}

// Build a 16-wide bf16 A/B fragment: two contiguous 8-element (16B) chunks
// at +0 and +16 elements (per CDNA5 16-bit A/B VGPR layout).
__device__ __forceinline__ v16bf mk16(const __bf16* p) {
  union { v16bf v; v8bf h[2]; } u;
  u.h[0] = *(const v8bf*)(p);
  u.h[1] = *(const v8bf*)(p + 16);
  return u.v;
}

__device__ __forceinline__ v8f wmma_bf16(v16bf a, v16bf b, v8f c) {
  return __builtin_amdgcn_wmma_f32_16x16x32_bf16(
      /*neg_a=*/false, a, /*neg_b=*/false, b,
      /*c_mod=*/(short)0, c, /*reuse_a=*/false, /*reuse_b=*/false);
}

// Async DMA: one 16B chunk global -> LDS (per-lane addresses, ASYNCcnt).
__device__ __forceinline__ void async_ld128(void* ldsPtr, const __bf16* g) {
  uint32_t off = (uint32_t)(size_t)ldsPtr;  // low 32 bits of generic = LDS offset
  asm volatile("global_load_async_to_lds_b128 %0, %1, off"
               :: "v"(off), "v"(g) : "memory");
}

// Cooperatively stage a 64x64 bf16 tile (src row stride srcStride elements)
// into LDS rows padded to 72 elements. 256 threads x 2 chunks of 16B.
__device__ __forceinline__ void stage64x64(__bf16* dst, const __bf16* src,
                                           int srcStride, int tid) {
#pragma unroll
  for (int r = 0; r < 2; ++r) {
    const int c = tid + r * 256;   // 0..511 chunk id
    const int row = c >> 3;
    const int col = (c & 7) << 3;
    async_ld128(dst + row * 72 + col, src + (size_t)row * srcStride + col);
  }
}

// ---------------- kernel 0a: hidden fp32 -> bf16 ----------------
__global__ void cvt_x_kernel(const float* __restrict__ x, __bf16* __restrict__ xb) {
  int i = (blockIdx.x * 256 + threadIdx.x) * 4;
  float4 f = *(const float4*)(x + i);
  union { __bf16 b[4]; uint2 u; } p;
  p.b[0] = f2bf(f.x); p.b[1] = f2bf(f.y); p.b[2] = f2bf(f.z); p.b[3] = f2bf(f.w);
  *(uint2*)(xb + i) = p.u;
}

// ---------------- kernel 0b: W fp32 [k][o] -> bf16 transposed Wt[o][k] ------
__global__ void cvt_w_kernel(const float* __restrict__ Wq,
                             const float* __restrict__ Wk,
                             const float* __restrict__ Wv,
                             __bf16* __restrict__ wt) {
  const float* W = (blockIdx.z == 0) ? Wq : (blockIdx.z == 1) ? Wk : Wv;
  __bf16* o = wt + (size_t)blockIdx.z * (DMsz * DMsz);
  int oc = blockIdx.x;
  int k  = threadIdx.x * 4;
  union { __bf16 b[4]; uint2 u; } p;
#pragma unroll
  for (int t = 0; t < 4; ++t)
    p.b[t] = f2bf(W[(size_t)(k + t) * DMsz + oc]);
  *(uint2*)(o + (size_t)oc * DMsz + k) = p.u;
}

// ---------------- kernel 1: fused QKV projection (bf16 WMMA) ----------------
__global__ void qkv_gemm_kernel(const __bf16* __restrict__ xb,
                                const __bf16* __restrict__ wt,
                                const float* __restrict__ bq,
                                const float* __restrict__ bk,
                                const float* __restrict__ bv,
                                __bf16* __restrict__ qo,
                                __bf16* __restrict__ ko,
                                __bf16* __restrict__ vo) {
  const int lane = threadIdx.x & 31, wave = threadIdx.x >> 5;
  const int hl = lane >> 4, lr = lane & 15;
  const int m0 = (blockIdx.x * 8 + wave) * 16;
  const int n0 = blockIdx.y * 64;
  const int mat = blockIdx.z;
  const __bf16* Bw = wt + (size_t)mat * (DMsz * DMsz);
  const float* bias = (mat == 0) ? bq : (mat == 1) ? bk : bv;

  v8f acc[4] = {};
  const __bf16* arow = xb + (size_t)(m0 + lr) * DMsz + hl * 8;

  for (int kt = 0; kt < DMsz; kt += 32) {
    if (kt + 32 < DMsz)
      __builtin_prefetch(arow + kt + 32, 0, 0);
    v16bf af = mk16(arow + kt);
#pragma unroll
    for (int j = 0; j < 4; ++j) {
      const __bf16* brow = Bw + (size_t)(n0 + 16 * j + lr) * DMsz + kt + hl * 8;
      v16bf bfr = mk16(brow);
      acc[j] = wmma_bf16(af, bfr, acc[j]);
    }
  }

#pragma unroll
  for (int j = 0; j < 4; ++j) {
    const int o = n0 + 16 * j + lr;
    const float bvl = bias[o];
    const int h = o >> 6, d = o & 63;
#pragma unroll
    for (int v = 0; v < 8; ++v) {
      const int row = m0 + v + 8 * hl;
      const int b = row >> 11, s = row & 2047;
      const float val = acc[j][v] + bvl;
      if (mat == 0) {
        qo[((size_t)(b * Hsz + h) * Ssz + s) * Dsz + d] = f2bf(val * 0.125f);
      } else if (mat == 1) {
        ko[((size_t)(b * Hsz + h) * Ssz + s) * Dsz + d] = f2bf(val);
      } else {
        vo[((size_t)(b * Hsz + h) * Dsz + d) * Ssz + s] = f2bf(val);
      }
    }
  }
}

// ---------------- kernel 2: attention (two-pass softmax + ctx) --------------
// grid (S/128, H, B), block 256 (8 waves, each wave owns 16 queries).
// K/V tiles (64 keys) staged into LDS via async DMA, double-buffered.
__global__ void attention_kernel(const __bf16* __restrict__ qbf,
                                 const __bf16* __restrict__ kbf,
                                 const __bf16* __restrict__ vtbf,
                                 const float* __restrict__ amask,
                                 float* __restrict__ ctx,
                                 float* __restrict__ probs) {
  __shared__ __align__(16) __bf16 Ksh[2][64][72];
  __shared__ __align__(16) __bf16 Vsh[2][64][72];
  __shared__ __align__(16) __bf16 pStage[8][16][40];

  const int tid = threadIdx.x;
  const int lane = tid & 31, wave = tid >> 5;
  const int hl = lane >> 4, lr = lane & 15;
  const int q0 = blockIdx.x * 128 + wave * 16;
  const int h = blockIdx.y, b = blockIdx.z;

  const __bf16* Qh = qbf + (size_t)(b * Hsz + h) * Ssz * Dsz;
  const __bf16* Kh = kbf + (size_t)(b * Hsz + h) * Ssz * Dsz;
  const __bf16* Vh = vtbf + (size_t)(b * Hsz + h) * Dsz * Ssz;
  const float* mask = amask + (size_t)b * Ssz;

  // Q fragments (head-dim 0..31 and 32..63) held in registers for both passes
  const __bf16* qrow = Qh + (size_t)(q0 + lr) * Dsz + hl * 8;
  const v16bf qf0 = mk16(qrow);
  const v16bf qf1 = mk16(qrow + 32);

  float m[8], l[8];
#pragma unroll
  for (int v = 0; v < 8; ++v) { m[v] = -1e30f; l[v] = 0.0f; }

  // ---------------- pass 1: online row max / sum-of-exp ----------------
  stage64x64(&Ksh[0][0][0], Kh, Dsz, tid);
  for (int t = 0; t < Ssz / 64; ++t) {
    const int kt0 = t * 64;
    const int buf = t & 1;
    if (t + 1 < Ssz / 64) {
      stage64x64(&Ksh[buf ^ 1][0][0], Kh + (size_t)(kt0 + 64) * Dsz, Dsz, tid);
      WAIT_ASYNCCNT(2);   // current tile's 2 DMAs landed; next 2 in flight
    } else {
      WAIT_ASYNCCNT(0);
    }
    __syncthreads();
#pragma unroll
    for (int ks = 0; ks < 4; ++ks) {
      const int ktl = ks * 16;
      v16bf kf0 = mk16(&Ksh[buf][ktl + lr][hl * 8]);
      v16bf kf1 = mk16(&Ksh[buf][ktl + lr][32 + hl * 8]);
      v8f c = {};
      c = wmma_bf16(qf0, kf0, c);
      c = wmma_bf16(qf1, kf1, c);
      const float mv = mask[kt0 + ktl + lr];
#pragma unroll
      for (int v = 0; v < 8; ++v) {
        float s = c[v] + mv;
        float x = s;
        x = fmaxf(x, __shfl_xor(x, 1));
        x = fmaxf(x, __shfl_xor(x, 2));
        x = fmaxf(x, __shfl_xor(x, 4));
        x = fmaxf(x, __shfl_xor(x, 8));
        const float mn = fmaxf(m[v], x);
        float e = exp2f((s - mn) * L2E);
        float sum = e;
        sum += __shfl_xor(sum, 1);
        sum += __shfl_xor(sum, 2);
        sum += __shfl_xor(sum, 4);
        sum += __shfl_xor(sum, 8);
        l[v] = l[v] * exp2f((m[v] - mn) * L2E) + sum;
        m[v] = mn;
      }
    }
    WAIT_DSCNT0;        // DS reads done before next tile overwrites buffer
    __syncthreads();
  }

  float rinv[8];
#pragma unroll
  for (int v = 0; v < 8; ++v) rinv[v] = 1.0f / l[v];

  // ------- pass 2: recompute scores, emit probs, accumulate ctx -------
  float* pr = probs + ((size_t)(b * Hsz + h) * Ssz + q0) * Ssz;
  v8f acc[4] = {};

  stage64x64(&Ksh[0][0][0], Kh, Dsz, tid);
  stage64x64(&Vsh[0][0][0], Vh, Ssz, tid);
  for (int t = 0; t < Ssz / 64; ++t) {
    const int kt0 = t * 64;
    const int buf = t & 1;
    if (t + 1 < Ssz / 64) {
      stage64x64(&Ksh[buf ^ 1][0][0], Kh + (size_t)(kt0 + 64) * Dsz, Dsz, tid);
      stage64x64(&Vsh[buf ^ 1][0][0], Vh + kt0 + 64, Ssz, tid);
      WAIT_ASYNCCNT(4);
    } else {
      WAIT_ASYNCCNT(0);
    }
    __syncthreads();
#pragma unroll
    for (int pair = 0; pair < 2; ++pair) {
      const int kl0 = pair * 32;
#pragma unroll
      for (int sub = 0; sub < 2; ++sub) {
        const int ktl = kl0 + sub * 16;
        v16bf kf0 = mk16(&Ksh[buf][ktl + lr][hl * 8]);
        v16bf kf1 = mk16(&Ksh[buf][ktl + lr][32 + hl * 8]);
        v8f c = {};
        c = wmma_bf16(qf0, kf0, c);
        c = wmma_bf16(qf1, kf1, c);
        const float mv = mask[kt0 + ktl + lr];
#pragma unroll
        for (int v = 0; v < 8; ++v) {
          const float p = exp2f((c[v] + mv - m[v]) * L2E) * rinv[v];
          pr[(size_t)(v + 8 * hl) * Ssz + kt0 + ktl + lr] = p;   // probs out
          pStage[wave][v + 8 * hl][sub * 16 + lr] = f2bf(p);     // stage P
        }
      }
      WAIT_DSCNT0;  // same-wave LDS turn-around for the C->A layout shuffle
      const v16bf pf = mk16(&pStage[wave][lr][hl * 8]);
#pragma unroll
      for (int j = 0; j < 4; ++j) {
        v16bf vf = mk16(&Vsh[buf][j * 16 + lr][kl0 + hl * 8]);
        acc[j] = wmma_bf16(pf, vf, acc[j]);
      }
    }
    WAIT_DSCNT0;
    __syncthreads();
  }

  // -------- store ctx [B,S,DM] --------
#pragma unroll
  for (int j = 0; j < 4; ++j) {
    const int col = h * Dsz + j * 16 + lr;
#pragma unroll
    for (int v = 0; v < 8; ++v) {
      const int row = q0 + v + 8 * hl;
      ctx[((size_t)b * Ssz + row) * DMsz + col] = acc[j][v];
    }
  }
}

// ---------------- host launch ----------------
extern "C" void kernel_launch(void* const* d_in, const int* in_sizes, int n_in,
                              void* d_out, int out_size, void* d_ws, size_t ws_size,
                              hipStream_t stream) {
  const float* hidden = (const float*)d_in[0];
  const float* amask  = (const float*)d_in[1];
  const float* Wq = (const float*)d_in[2];
  const float* bq = (const float*)d_in[3];
  const float* Wk = (const float*)d_in[4];
  const float* bk = (const float*)d_in[5];
  const float* Wv = (const float*)d_in[6];
  const float* bv = (const float*)d_in[7];

  float* out = (float*)d_out;
  float* ctx = out;                                 // B*S*DM floats
  float* probs = out + (size_t)Bsz * Ssz * DMsz;    // B*H*S*S floats

  __bf16* xb = (__bf16*)d_ws;                       // 8192x1024
  __bf16* wt = xb + (size_t)Bsz * Ssz * DMsz;       // 3x 1024x1024 (transposed)
  __bf16* qo = wt + (size_t)3 * DMsz * DMsz;        // [B,H,S,D] (pre-scaled)
  __bf16* ko = qo + (size_t)Bsz * Ssz * DMsz;       // [B,H,S,D]
  __bf16* vo = ko + (size_t)Bsz * Ssz * DMsz;       // [B,H,D,S]

  cvt_x_kernel<<<(Bsz * Ssz * DMsz) / 1024, 256, 0, stream>>>(hidden, xb);
  cvt_w_kernel<<<dim3(DMsz, 1, 3), 256, 0, stream>>>(Wq, Wk, Wv, wt);
  qkv_gemm_kernel<<<dim3((Bsz * Ssz) / 128, DMsz / 64, 3), 256, 0, stream>>>(
      xb, wt, bq, bk, bv, qo, ko, vo);
  attention_kernel<<<dim3(Ssz / 128, Hsz, Bsz), 256, 0, stream>>>(
      qo, ko, vo, amask, ctx, probs);
}
